// GeometricLMv3_62955630625402
// MI455X (gfx1250) — matrix operand order repaired and checked by reading, not compiled, not run on hardware
//
#include <hip/hip_runtime.h>
#include <hip/hip_bf16.h>

// GeometricLMv3 for MI455X (gfx1250, wave32).
// Pipeline:
//   K0 qproj:   qp = Wc^T @ query, c0 = bc.query           (algebraic collapse of attention prologue)
//   K1 context: attention-pooled context[B,E]
//   K2 neuron:  act[b,n]=RBF, A[b, n*H+h] = act*relu(ctx@Wh)  -> bf16 workspace
//   K3 gemm:    logits[64,32000] = A[64x4096] @ W[4096x32000] (+ act@bias epilogue)
//               bf16 WMMA (v_wmma_f32_16x16x32_bf16), f32 accumulate; W streamed with
//               non-temporal loads (524MB read-once; keep L2 for the hot 512KB A) and
//               converted f32->bf16 in-register. Memory-bound: ~524MB / 23.3TB/s.

typedef __bf16 bf16_t;
typedef __attribute__((ext_vector_type(16))) __bf16 v16bf;
typedef __attribute__((ext_vector_type(8)))  __bf16 v8bf;
typedef __attribute__((ext_vector_type(8)))  float  v8f;

#define VOCAB 32000
#define EDIM  256
#define HDIM  64
#define SEQ   64
#define NNEUR 64
#define BATCH 64
#define KDIM  (NNEUR * HDIM)   // 4096

// ---------------------------------------------------------------------------
// WMMA fragment loaders (wave32, 16x16x32 bf16).
// B fragment: 32(K) x 16(N). Lane n = lane&15 selects column. Lanes 0-15 hold
// K=0..15, lanes 16-31 hold K=16..31; VGPR j holds K = 2j, 2j+1 (packed).
// Source is f32 (row stride ldv), streamed with TH=NT, converted in-register.
__device__ __forceinline__ v16bf load_bfrag_f32col_nt(const float* __restrict__ colbase,
                                                      long ldv, int hk) {
  const float* p = colbase + (long)hk * 16 * ldv;
  float x[16];
#pragma unroll
  for (int j = 0; j < 16; ++j) x[j] = __builtin_nontemporal_load(p + (long)j * ldv);
  v16bf b;
#pragma unroll
  for (int j = 0; j < 16; ++j) b[j] = (bf16_t)x[j];
  return b;
}

// A fragment: 16(M) x 32(K), bf16 source, row-major stride lda (elements).
// Lane m = lane&15; half hk = lane>>4. VGPR 0-3: K = hk*16 + 0..7,
// VGPR 4-7: K = hk*16 + 8..15 (contiguous, matches B fragment).
__device__ __forceinline__ v16bf load_afrag_bf16(const bf16_t* __restrict__ tile,
                                                 int lda, int lane) {
  const int m = lane & 15, hk = lane >> 4;
  const bf16_t* p = tile + (long)m * lda + hk * 16;
  v8bf lo = *(const v8bf*)(p);        // 16B aligned: lda multiple of 8
  v8bf hi = *(const v8bf*)(p + 8);
  v16bf a;
#pragma unroll
  for (int i = 0; i < 8; ++i) { a[i] = lo[i]; a[8 + i] = hi[i]; }
  return a;
}

// ---------------------------------------------------------------------------
// K0: qp[e] = sum_i query[i]*Wc[i,e];  c0 = sum_i query[i]*bc[i]
__global__ __launch_bounds__(256) void qproj_kernel(
    const float* __restrict__ Wc, const float* __restrict__ bc,
    const float* __restrict__ query, float* __restrict__ qp, float* __restrict__ c0) {
  __shared__ float red[256];
  const int t = threadIdx.x;
  red[t] = query[t] * bc[t];
  __syncthreads();
#pragma unroll
  for (int s = 128; s > 0; s >>= 1) {
    if (t < s) red[t] += red[t + s];
    __syncthreads();
  }
  if (t == 0) *c0 = red[0];
  float acc = 0.f;
  for (int i = 0; i < EDIM; ++i) acc += query[i] * Wc[i * EDIM + t];  // coalesced over t
  qp[t] = acc;
}

// ---------------------------------------------------------------------------
// K1: one block per batch row. Gather+scores via LDS chunks, softmax, pooled context.
__global__ __launch_bounds__(256) void context_kernel(
    const int* __restrict__ token_ids, const float* __restrict__ token_embed,
    const float* __restrict__ pos_embed, const float* __restrict__ qp,
    const float* __restrict__ c0p, float* __restrict__ ctx) {
  __shared__ float emb[SEQ][68];     // 64-col chunk of embeds, padded
  __shared__ float qpl[EDIM];
  __shared__ int   toks[SEQ];
  __shared__ float sc[SEQ], esc[SEQ], attn[SEQ];
  __shared__ float red4[4][68];

  const int b = blockIdx.x;
  const int t = threadIdx.x;
  qpl[t] = qp[t];
  if (t < SEQ) toks[t] = token_ids[b * SEQ + t];
  __syncthreads();

  // scores: dot(embeds[s,:], qp) accumulated over 4 column chunks
  float dacc = 0.f;
  for (int ch = 0; ch < 4; ++ch) {
    for (int idx = t; idx < SEQ * 64; idx += 256) {
      int s = idx >> 6, e = idx & 63;
      emb[s][e] = token_embed[(long)toks[s] * EDIM + ch * 64 + e] +
                  pos_embed[s * EDIM + ch * 64 + e];
    }
    __syncthreads();
    if (t < SEQ) {
#pragma unroll
      for (int e = 0; e < 64; ++e) dacc += emb[t][e] * qpl[ch * 64 + e];
    }
    __syncthreads();
  }
  if (t < SEQ) sc[t] = (dacc + *c0p) * 0.0625f;  // 1/sqrt(256)
  __syncthreads();
  if (t < SEQ) {
    float m = -1e30f;
    for (int s = 0; s < SEQ; ++s) m = fmaxf(m, sc[s]);
    esc[t] = __expf(sc[t] - m);
  }
  __syncthreads();
  if (t < SEQ) {
    float sum = 0.f;
    for (int s = 0; s < SEQ; ++s) sum += esc[s];
    attn[t] = esc[t] / sum;
  }
  __syncthreads();

  // context[b, :] = sum_s attn[s] * embeds[s, :]
  const int e_loc = t & 63, sg = t >> 6;
  for (int ch = 0; ch < 4; ++ch) {
    for (int idx = t; idx < SEQ * 64; idx += 256) {
      int s = idx >> 6, e = idx & 63;
      emb[s][e] = token_embed[(long)toks[s] * EDIM + ch * 64 + e] +
                  pos_embed[s * EDIM + ch * 64 + e];
    }
    __syncthreads();
    float cp = 0.f;
#pragma unroll
    for (int i = 0; i < 16; ++i) cp += attn[sg * 16 + i] * emb[sg * 16 + i][e_loc];
    red4[sg][e_loc] = cp;
    __syncthreads();
    if (t < 64) ctx[b * EDIM + ch * 64 + t] =
        red4[0][t] + red4[1][t] + red4[2][t] + red4[3][t];
    __syncthreads();
  }
}

// ---------------------------------------------------------------------------
// K2: one block per neuron n. act[b,n]=exp(-||ctx-pos||^2/(2 BW^2));
// A[b, n*H+h] = bf16(act[b,n] * relu(sum_e ctx[b,e]*Wh[e,h]))
__global__ __launch_bounds__(256) void neuron_kernel(
    const float* __restrict__ ctx, const float* __restrict__ positions,
    const float* __restrict__ neuron_hidden, bf16_t* __restrict__ Abf,
    bf16_t* __restrict__ actbf) {
  __shared__ float posl[EDIM];
  __shared__ float actl[BATCH];
  const int n = blockIdx.x;
  const int t = threadIdx.x;
  posl[t] = positions[n * EDIM + t];
  __syncthreads();
  if (t < BATCH) {
    float d2 = 0.f;
    for (int e = 0; e < EDIM; ++e) {
      float d = ctx[t * EDIM + e] - posl[e];
      d2 += d * d;
    }
    float a = __expf(-d2 * (1.0f / (2.0f * 0.3f * 0.3f)));
    actl[t] = a;
    actbf[t * NNEUR + n] = (bf16_t)a;
  }
  __syncthreads();

  const int h = t & 63, bq = t >> 6;  // thread -> hidden unit h, batch quad bq
  const float* Wh = neuron_hidden + (long)n * EDIM * HDIM;  // [256][64]
  float accv[16];
#pragma unroll
  for (int i = 0; i < 16; ++i) accv[i] = 0.f;
  for (int e = 0; e < EDIM; e += 4) {
    float w0 = Wh[(e + 0) * HDIM + h];
    float w1 = Wh[(e + 1) * HDIM + h];
    float w2 = Wh[(e + 2) * HDIM + h];
    float w3 = Wh[(e + 3) * HDIM + h];
#pragma unroll
    for (int i = 0; i < 16; ++i) {
      const float4 c4 = *(const float4*)&ctx[(bq * 16 + i) * EDIM + e];
      accv[i] += c4.x * w0 + c4.y * w1 + c4.z * w2 + c4.w * w3;
    }
  }
#pragma unroll
  for (int i = 0; i < 16; ++i) {
    int b = bq * 16 + i;
    float hv = fmaxf(accv[i], 0.f);
    Abf[(long)b * KDIM + n * HDIM + h] = (bf16_t)(actl[b] * hv);
  }
}

// ---------------------------------------------------------------------------
// K3: logits[64,32000] = A[64x4096]@W[4096x32000] + act[64x64]@bias[64x32000]
// Block: 128 V-columns, 8 waves; wave owns 16 columns x full M=64 (4 WMMA M-tiles).
// A fragments issued first (single overlapped wait), W streamed NT, output stored NT.
__global__ __launch_bounds__(256) void geo_gemm_kernel(
    const bf16_t* __restrict__ Abf, const bf16_t* __restrict__ actbf,
    const float* __restrict__ W, const float* __restrict__ bias,
    float* __restrict__ out) {
  const int lane = threadIdx.x & 31;
  const int wave = threadIdx.x >> 5;
  const int hk   = lane >> 4;
  const int v    = blockIdx.x * 128 + wave * 16 + (lane & 15);

  v8f acc[4];
#pragma unroll
  for (int mt = 0; mt < 4; ++mt)
#pragma unroll
    for (int r = 0; r < 8; ++r) acc[mt][r] = 0.f;

  const float* wcol = W + v;
#pragma unroll 2
  for (int ks = 0; ks < KDIM; ks += 32) {
    // All loads for this K-step in flight before any consumer:
    v16bf af[4];
#pragma unroll
    for (int mt = 0; mt < 4; ++mt)
      af[mt] = load_afrag_bf16(Abf + (long)(mt * 16) * KDIM + ks, KDIM, lane);
    v16bf bf = load_bfrag_f32col_nt(wcol + (long)ks * VOCAB, VOCAB, hk);
    if (ks + 64 < KDIM)  // pull next-next K-slab toward GL2 (global_prefetch_b8)
      __builtin_prefetch(wcol + (long)(ks + 64 + hk * 16) * VOCAB, 0, 1);
#pragma unroll
    for (int mt = 0; mt < 4; ++mt)
      acc[mt] = __builtin_amdgcn_wmma_f32_16x16x32_bf16(
          false, af[mt], false, bf, (short)0, acc[mt], false, false);
  }

  // bias epilogue: logits += act[64x64] @ neuron_bias[64x32000] (2 K-steps)
  const float* bcol = bias + v;
#pragma unroll
  for (int ks = 0; ks < NNEUR; ks += 32) {
    v16bf af[4];
#pragma unroll
    for (int mt = 0; mt < 4; ++mt)
      af[mt] = load_afrag_bf16(actbf + (mt * 16) * NNEUR + ks, NNEUR, lane);
    v16bf bf = load_bfrag_f32col_nt(bcol + (long)ks * VOCAB, VOCAB, hk);
#pragma unroll
    for (int mt = 0; mt < 4; ++mt)
      acc[mt] = __builtin_amdgcn_wmma_f32_16x16x32_bf16(
          false, af[mt], false, bf, (short)0, acc[mt], false, false);
  }

  // C/D layout: VGPR r -> row mt*16 + hk*8 + r; column v. Covers M=0..63 exactly.
  // Output is write-once: bypass cache with NT stores.
#pragma unroll
  for (int mt = 0; mt < 4; ++mt)
#pragma unroll
    for (int r = 0; r < 8; ++r)
      __builtin_nontemporal_store(acc[mt][r],
                                  &out[(long)(mt * 16 + hk * 8 + r) * VOCAB + v]);
}

// ---------------------------------------------------------------------------
extern "C" void kernel_launch(void* const* d_in, const int* in_sizes, int n_in,
                              void* d_out, int out_size, void* d_ws, size_t ws_size,
                              hipStream_t stream) {
  const int*   token_ids     = (const int*)  d_in[0];
  const float* token_embed   = (const float*)d_in[1];
  const float* pos_embed     = (const float*)d_in[2];
  const float* Wc            = (const float*)d_in[3];
  const float* bc            = (const float*)d_in[4];
  const float* query         = (const float*)d_in[5];
  const float* positions     = (const float*)d_in[6];
  const float* neuron_hidden = (const float*)d_in[7];
  const float* neuron_output = (const float*)d_in[8];
  const float* neuron_bias   = (const float*)d_in[9];
  float* out = (float*)d_out;

  char* ws = (char*)d_ws;
  float*  qp    = (float*)(ws + 0);                     //   1 KB : qp[256]
  float*  c0    = (float*)(ws + 1024);                  //   4 B  : scalar
  float*  ctx   = (float*)(ws + 4096);                  //  64 KB : ctx[64][256] f32
  bf16_t* actbf = (bf16_t*)(ws + 4096 + 65536);         //   8 KB : act[64][64] bf16
  bf16_t* Abf   = (bf16_t*)(ws + 4096 + 65536 + 8192);  // 512 KB : A[64][4096] bf16

  qproj_kernel  <<<1,          256, 0, stream>>>(Wc, bc, query, qp, c0);
  context_kernel<<<BATCH,      256, 0, stream>>>(token_ids, token_embed, pos_embed,
                                                 qp, c0, ctx);
  neuron_kernel <<<NNEUR,      256, 0, stream>>>(ctx, positions, neuron_hidden,
                                                 Abf, actbf);
  geo_gemm_kernel<<<VOCAB/128, 256, 0, stream>>>(Abf, actbf, neuron_output,
                                                 neuron_bias, out);
}
